// Repro_85590108274908
// MI455X (gfx1250) — matrix operand order repaired and checked by reading, not compiled, not run on hardware
//
#include <hip/hip_runtime.h>

// ---------------------------------------------------------------------------
// Fused embedding-gather + 4x Linear(400->400) attention/gating network for
// MI455X (gfx1250).  Math in f16 WMMA (v_wmma_f32_16x16x32_f16) with f32
// accumulation; gathers/outputs are the HBM-bound part (~1.7 GB @ 23.3 TB/s).
// ---------------------------------------------------------------------------

#define B_ROWS 131072
#define DD     400
#define KP     416            // K padded to multiple of 32 (13 wmma k-steps)
#define STR    424            // LDS row stride in halves (16B aligned, bank skew)
#define MT     64             // batch rows per block
#define NW     8              // waves per block (256 threads, wave32)
#define NTILES 25             // 400 / 16 output column tiles
#define WELEM  (DD * KP)      // elements per converted weight matrix

typedef __attribute__((ext_vector_type(16))) _Float16 v16h;
typedef __attribute__((ext_vector_type(8)))  _Float16 v8h;
typedef __attribute__((ext_vector_type(8)))  float    v8f;

// LDS region offsets (in halves)
#define R_E0 0
#define R_E1 (MT * STR)
#define R_H0 (2 * MT * STR)
#define R_H1 (3 * MT * STR)

// A-operand (16x32 f16, ISA 7.12.2): lane m=lane&15, lh=lane>>4 holds
// K = {base+lh*8 .. +7} and {base+16+lh*8 .. +7}  -> two 16B reads.
__device__ __forceinline__ v16h ld_a(const _Float16* p, int lh) {
  v8h lo = *(const v8h*)(p + lh * 8);
  v8h hi = *(const v8h*)(p + 16 + lh * 8);
  return __builtin_shufflevector(lo, hi, 0,1,2,3,4,5,6,7,8,9,10,11,12,13,14,15);
}

// B-operand (32x16 f16): lane n=lane&15 holds K = {lh*8 .. +7} and
// {16+lh*8 .. +7} (same K order as A) -> two 16B reads.
__device__ __forceinline__ v16h ld_b(const _Float16* p, int lh) {
  v8h lo = *(const v8h*)(p + lh * 8);
  v8h hi = *(const v8h*)(p + 16 + lh * 8);
  return __builtin_shufflevector(lo, hi, 0,1,2,3,4,5,6,7,8,9,10,11,12,13,14,15);
}

__device__ __forceinline__ void mm2(const _Float16* A0, const _Float16* A1,
                                    const _Float16* W, int lh,
                                    v8f& acc0, v8f& acc1) {
  for (int kk = 0; kk < KP; kk += 32) {
    v16h b  = ld_b(W + kk, lh);                 // shared across both branches
    v16h x0 = ld_a(A0 + kk, lh);
    acc0 = __builtin_amdgcn_wmma_f32_16x16x32_f16(false, x0, false, b,
                                                  (short)0, acc0, false, false);
    v16h x1 = ld_a(A1 + kk, lh);
    acc1 = __builtin_amdgcn_wmma_f32_16x16x32_f16(false, x1, false, b,
                                                  (short)0, acc1, false, false);
  }
}

__device__ __forceinline__ void mm1(const _Float16* A0, const _Float16* W,
                                    int lh, v8f& acc) {
  for (int kk = 0; kk < KP; kk += 32) {
    v16h b = ld_b(W + kk, lh);
    v16h x = ld_a(A0 + kk, lh);
    acc = __builtin_amdgcn_wmma_f32_16x16x32_f16(false, x, false, b,
                                                 (short)0, acc, false, false);
  }
}

// Convert the four [400x400] f32 weight matrices to f16 [400x416] (zero pad).
__global__ void cvt_weights(const float* __restrict__ w3,
                            const float* __restrict__ w5,
                            const float* __restrict__ w7,
                            const float* __restrict__ w9,
                            _Float16* __restrict__ out) {
  int t = blockIdx.x * blockDim.x + threadIdx.x;
  if (t >= 4 * WELEM) return;
  int m = t / WELEM, e = t - m * WELEM;
  int row = e / KP, col = e - row * KP;
  const float* src = (m == 0) ? w3 : (m == 1) ? w5 : (m == 2) ? w7 : w9;
  out[t] = (col < DD) ? (_Float16)src[row * DD + col] : (_Float16)0.f;
}

__global__ __launch_bounds__(256, 1) void fused_fwd(
    const float* __restrict__ a0, const float* __restrict__ a1,
    const float* __restrict__ a2,
    const float* __restrict__ b4, const float* __restrict__ b6,
    const float* __restrict__ b8, const float* __restrict__ b10,
    const int* __restrict__ idx,
    const _Float16* __restrict__ wh,
    float* __restrict__ out) {
  __shared__ _Float16 smem[4 * MT * STR];   // 217 KB (<320 KB/WGP)

  const int tid  = threadIdx.x;
  const int wid  = tid >> 5;
  const int lane = tid & 31;
  const int lh   = lane >> 4;
  const int lm   = lane & 15;
  const int row0 = blockIdx.x * MT;

  // ---- Phase A: gather e0 = a0[i0]+a1[i1], e1 = a0[i2]+a1[i3] -> f16 LDS ----
  for (int r = wid; r < MT; r += NW) {
    const int4 iv = ((const int4*)idx)[row0 + r];
    const float* p00 = a0 + (size_t)iv.x * DD;
    const float* p01 = a1 + (size_t)iv.y * DD;
    const float* p10 = a0 + (size_t)iv.z * DD;
    const float* p11 = a1 + (size_t)iv.w * DD;
    for (int d = lane; d < KP; d += 32) {
      float v0 = 0.f, v1 = 0.f;
      if (d < DD) { v0 = p00[d] + p01[d]; v1 = p10[d] + p11[d]; }
      smem[R_E0 + r * STR + d] = (_Float16)v0;
      smem[R_E1 + r * STR + d] = (_Float16)v1;
      if (d >= DD) {                       // zero K-pad of h regions too
        smem[R_H0 + r * STR + d] = (_Float16)0.f;
        smem[R_H1 + r * STR + d] = (_Float16)0.f;
      }
    }
  }
  __syncthreads();

  // ---- GEMM1: h[k] = relu(e[k] @ W3^T + b4) -> f16 LDS ----
  for (int it = wid; it < (MT / 16) * NTILES; it += NW) {
    int mt = it / NTILES, nt = it - mt * NTILES;
    int n  = nt * 16 + lm;
    const _Float16* Wp = wh + (size_t)n * KP;
    const _Float16* A0 = &smem[R_E0 + (mt * 16 + lm) * STR];
    const _Float16* A1 = &smem[R_E1 + (mt * 16 + lm) * STR];
    v8f acc0 = {}, acc1 = {};
    mm2(A0, A1, Wp, lh, acc0, acc1);
    float bn = b4[n];
#pragma unroll
    for (int e = 0; e < 8; e++) {
      int row = mt * 16 + e + 8 * lh;
      smem[R_H0 + row * STR + n] = (_Float16)fmaxf(acc0[e] + bn, 0.f);
      smem[R_H1 + row * STR + n] = (_Float16)fmaxf(acc1[e] + bn, 0.f);
    }
  }
  __syncthreads();

  // ---- GEMM2: scores = h @ W5^T + b6; 2-way softmax pool of e -> out[0] ----
  for (int it = wid; it < (MT / 16) * NTILES; it += NW) {
    int mt = it / NTILES, nt = it - mt * NTILES;
    int n  = nt * 16 + lm;
    const _Float16* Wp = wh + (size_t)WELEM + (size_t)n * KP;
    const _Float16* A0 = &smem[R_H0 + (mt * 16 + lm) * STR];
    const _Float16* A1 = &smem[R_H1 + (mt * 16 + lm) * STR];
    v8f acc0 = {}, acc1 = {};
    mm2(A0, A1, Wp, lh, acc0, acc1);
    float bn = b6[n];
#pragma unroll
    for (int e = 0; e < 8; e++) {
      int row = mt * 16 + e + 8 * lh;
      float s0 = acc0[e] + bn, s1 = acc1[e] + bn;
      float mx = fmaxf(s0, s1);
      float w0 = __expf(s0 - mx), w1 = __expf(s1 - mx);
      float e0v = (float)smem[R_E0 + row * STR + n];
      float e1v = (float)smem[R_E1 + row * STR + n];
      out[(size_t)(row0 + row) * DD + n] = (w0 * e0v + w1 * e1v) / (w0 + w1);
    }
  }
  __syncthreads();

  // ---- Phase D: gather v5[k] = a2[idx1], a2[idx3] into H regions ----
  for (int r = wid; r < MT; r += NW) {
    const int4 iv = ((const int4*)idx)[row0 + r];
    const float* q0 = a2 + (size_t)iv.y * DD;
    const float* q1 = a2 + (size_t)iv.w * DD;
    for (int d = lane; d < KP; d += 32) {
      float v0 = (d < DD) ? q0[d] : 0.f;
      float v1 = (d < DD) ? q1[d] : 0.f;
      smem[R_H0 + r * STR + d] = (_Float16)v0;
      smem[R_H1 + r * STR + d] = (_Float16)v1;
      if (d >= DD) smem[R_E0 + r * STR + d] = (_Float16)0.f;  // mean K-pad
    }
  }
  __syncthreads();

  // ---- GEMM3: m = mean_k relu(v5[k] @ W7^T + b8) -> f16 LDS (E0) ----
  for (int it = wid; it < (MT / 16) * NTILES; it += NW) {
    int mt = it / NTILES, nt = it - mt * NTILES;
    int n  = nt * 16 + lm;
    const _Float16* Wp = wh + (size_t)(2 * WELEM) + (size_t)n * KP;
    const _Float16* A0 = &smem[R_H0 + (mt * 16 + lm) * STR];
    const _Float16* A1 = &smem[R_H1 + (mt * 16 + lm) * STR];
    v8f acc0 = {}, acc1 = {};
    mm2(A0, A1, Wp, lh, acc0, acc1);
    float bn = b8[n];
#pragma unroll
    for (int e = 0; e < 8; e++) {
      int row = mt * 16 + e + 8 * lh;
      float mv = 0.5f * (fmaxf(acc0[e] + bn, 0.f) + fmaxf(acc1[e] + bn, 0.f));
      smem[R_E0 + row * STR + n] = (_Float16)mv;
    }
  }
  __syncthreads();

  // ---- GEMM4: gate = sigmoid(m @ W9^T + b10); out[1] = min(v5) * gate ----
  for (int it = wid; it < (MT / 16) * NTILES; it += NW) {
    int mt = it / NTILES, nt = it - mt * NTILES;
    int n  = nt * 16 + lm;
    const _Float16* Wp = wh + (size_t)(3 * WELEM) + (size_t)n * KP;
    const _Float16* A0 = &smem[R_E0 + (mt * 16 + lm) * STR];
    v8f acc = {};
    mm1(A0, Wp, lh, acc);
    float bn = b10[n];
#pragma unroll
    for (int e = 0; e < 8; e++) {
      int row = mt * 16 + e + 8 * lh;
      float g  = 1.f / (1.f + __expf(-(acc[e] + bn)));
      float m0 = (float)smem[R_H0 + row * STR + n];
      float m1 = (float)smem[R_H1 + row * STR + n];
      out[(size_t)B_ROWS * DD + (size_t)(row0 + row) * DD + n] = fminf(m0, m1) * g;
    }
  }
}

extern "C" void kernel_launch(void* const* d_in, const int* in_sizes, int n_in,
                              void* d_out, int out_size, void* d_ws, size_t ws_size,
                              hipStream_t stream) {
  const float* a0  = (const float*)d_in[0];
  const float* a1  = (const float*)d_in[1];
  const float* a2  = (const float*)d_in[2];
  const float* w3  = (const float*)d_in[3];
  const float* b4  = (const float*)d_in[4];
  const float* w5  = (const float*)d_in[5];
  const float* b6  = (const float*)d_in[6];
  const float* w7  = (const float*)d_in[7];
  const float* b8  = (const float*)d_in[8];
  const float* w9  = (const float*)d_in[9];
  const float* b10 = (const float*)d_in[10];
  const int*   idx = (const int*)d_in[12];
  float* out = (float*)d_out;
  _Float16* wh = (_Float16*)d_ws;   // 4 * 400 * 416 * 2 B = 1.27 MB scratch

  int cvt_threads = 4 * WELEM;
  cvt_weights<<<(cvt_threads + 255) / 256, 256, 0, stream>>>(w3, w5, w7, w9, wh);
  fused_fwd<<<B_ROWS / MT, 256, 0, stream>>>(a0, a1, a2, b4, b6, b8, b10,
                                             idx, wh, out);
}
